// BoundaryBranch_41042707481012
// MI455X (gfx1250) — matrix operand order repaired and checked by reading, not compiled, not run on hardware
//
#include <hip/hip_runtime.h>
#include <hip/hip_bf16.h>
#include <math.h>

typedef __attribute__((ext_vector_type(16))) _Float16 v16h;
typedef __attribute__((ext_vector_type(8)))  float    v8f;

#define KVOL 3136   // 64 * 7 * 7

union FragH { v16h h; uint4 u[2]; };
union AccF  { v8f  v; float e[8]; };

// ---------------------------------------------------------------------------
// Kernel 1: 7x7 stride-8 pad-3 conv (64->64) as implicit GEMM with WMMA f16.
// M = 1024 (b,oh,ow), N = 64 (oc), K = 3136 (ic*49). One block = 64x64 tile.
// 8 waves: wave w -> M-row group (w&3), N-half ((w>>2)*32), 2 WMMA tiles each.
// ---------------------------------------------------------------------------
__global__ __launch_bounds__(256)
void conv7x7s8_wmma(const float* __restrict__ x,      // (4,64,128,128)
                    const float* __restrict__ w,      // (64,64,7,7) OIHW
                    const float* __restrict__ bias,   // (64)
                    float* __restrict__ y)            // (4,64,16,16)
{
    __shared__ __align__(16) _Float16 Alds[64][32];   // A[m][k]
    __shared__ __align__(16) _Float16 Blds[64][32];   // B[n][k]

    const int tid   = threadIdx.x;
    const int lane  = tid & 31;          // wave32
    const int wv    = tid >> 5;
    const int mBase = blockIdx.x * 64;

    // staging assignment: 8 f16 elements per thread per tile
    const int a_m  = tid >> 2;           // 0..63
    const int a_k  = (tid & 3) * 8;      // 0,8,16,24
    const int am_g = mBase + a_m;
    const int ab   = am_g >> 8;
    const int aoh  = (am_g >> 4) & 15;
    const int aow  = am_g & 15;

    // fragment assignment
    const int mr    = wv & 3;
    const int nBase = (wv >> 2) * 32;
    const int fm    = mr * 16 + (lane & 15);
    const int khalf = lane >> 4;
    const int fn    = nBase + (lane & 15);

    v8f acc0 = {};
    v8f acc1 = {};

    for (int kk = 0; kk < KVOL; kk += 32) {
        __syncthreads();
        // ---- stage A tile: im2col gather with zero padding ----
        #pragma unroll
        for (int u = 0; u < 8; ++u) {
            int k  = kk + a_k + u;
            int ic = k / 49;
            int r  = k - ic * 49;
            int kh = r / 7;
            int kw = r - kh * 7;
            int ih = aoh * 8 - 3 + kh;
            int iw = aow * 8 - 3 + kw;
            float v = 0.f;
            if ((unsigned)ih < 128u && (unsigned)iw < 128u)
                v = x[(((ab << 6) + ic) << 14) + (ih << 7) + iw];
            Alds[a_m][a_k + u] = (_Float16)v;
        }
        // ---- stage B tile: OIHW weights are contiguous in k per oc ----
        #pragma unroll
        for (int u = 0; u < 8; ++u)
            Blds[a_m][a_k + u] = (_Float16)w[a_m * KVOL + kk + a_k + u];
        __syncthreads();

        // ---- load fragments, contiguous k order k = khalf*16 + s (same for A and B) ----
        FragH af, bf0, bf1;
        af.u[0]  = *(const uint4*)&Alds[fm][khalf * 16];       // K = kh*16..+7
        af.u[1]  = *(const uint4*)&Alds[fm][khalf * 16 + 8];   // K = kh*16+8..+15
        bf0.u[0] = *(const uint4*)&Blds[fn][khalf * 16];       // K = kh*16..+7
        bf0.u[1] = *(const uint4*)&Blds[fn][khalf * 16 + 8];
        bf1.u[0] = *(const uint4*)&Blds[fn + 16][khalf * 16];
        bf1.u[1] = *(const uint4*)&Blds[fn + 16][khalf * 16 + 8];

        acc0 = __builtin_amdgcn_wmma_f32_16x16x32_f16(false, af.h, false, bf0.h,
                                                      (short)0, acc0, false, false);
        acc1 = __builtin_amdgcn_wmma_f32_16x16x32_f16(false, af.h, false, bf1.h,
                                                      (short)0, acc1, false, false);
    }

    // ---- write C: VGPR v -> M = v + (lane>>4)*8, N = lane&15 ----
    AccF A0, A1; A0.v = acc0; A1.v = acc1;
    #pragma unroll
    for (int v = 0; v < 8; ++v) {
        int m   = mBase + mr * 16 + (lane >> 4) * 8 + v;
        int b   = m >> 8;
        int hw  = m & 255;
        int oc0 = nBase + (lane & 15);
        y[(((b << 6) + oc0) << 8) + hw]      = A0.e[v] + bias[oc0];
        y[(((b << 6) + oc0 + 16) << 8) + hw] = A1.e[v] + bias[oc0 + 16];
    }
}

// ---------------------------------------------------------------------------
// Kernel 2: per-channel mean / rsqrt(var+eps) over (N=4, H=16, W=16).
// ---------------------------------------------------------------------------
__global__ __launch_bounds__(256)
void bn_stats(const float* __restrict__ y, float* __restrict__ mean,
              float* __restrict__ rstd)
{
    __shared__ float ss[256];
    __shared__ float sq[256];
    const int c = blockIdx.x;
    const int t = threadIdx.x;
    float s = 0.f, q = 0.f;
    for (int b = 0; b < 4; ++b) {
        float v = y[(((b << 6) + c) << 8) + t];
        s += v; q += v * v;
    }
    ss[t] = s; sq[t] = q;
    __syncthreads();
    for (int o = 128; o > 0; o >>= 1) {
        if (t < o) { ss[t] += ss[t + o]; sq[t] += sq[t + o]; }
        __syncthreads();
    }
    if (t == 0) {
        float m   = ss[0] * (1.f / 1024.f);
        float var = sq[0] * (1.f / 1024.f) - m * m;
        mean[c] = m;
        rstd[c] = rsqrtf(var + 1e-5f);
    }
}

// ---------------------------------------------------------------------------
// Kernel 3: BN affine + ReLU + 1x1 conv + bias + ReLU.
// ---------------------------------------------------------------------------
template <int OC>
__global__ __launch_bounds__(256)
void bn_relu_conv1x1(const float* __restrict__ y, const float* __restrict__ mean,
                     const float* __restrict__ rstd, const float* __restrict__ gamma,
                     const float* __restrict__ beta, const float* __restrict__ w2,
                     const float* __restrict__ b2, float* __restrict__ out)
{
    const int p  = blockIdx.x * 256 + threadIdx.x;   // 0..1023
    const int b  = p >> 8;
    const int hw = p & 255;
    float acc[OC];
    #pragma unroll
    for (int o = 0; o < OC; ++o) acc[o] = b2[o];
    for (int c = 0; c < 64; ++c) {
        float v = y[(((b << 6) + c) << 8) + hw];
        v = fmaf((v - mean[c]) * rstd[c], gamma[c], beta[c]);
        v = fmaxf(v, 0.f);
        #pragma unroll
        for (int o = 0; o < OC; ++o) acc[o] = fmaf(v, w2[o * 64 + c], acc[o]);
    }
    #pragma unroll
    for (int o = 0; o < OC; ++o)
        out[((b * OC + o) << 8) + hw] = fmaxf(acc[o], 0.f);
}

// ---------------------------------------------------------------------------
// Kernel 4: zero the global bitmask.
// ---------------------------------------------------------------------------
__global__ void zero_u32(unsigned* p, int n)
{
    int i = blockIdx.x * 256 + threadIdx.x;
    if (i < n) p[i] = 0;
}

// ---------------------------------------------------------------------------
// Kernel 5: per-cell Fourier contour raster + parity in/out fill, entirely in
// LDS bit-planes (128x128 bits = 512 u32), OR-accumulated into global bitmask.
// ---------------------------------------------------------------------------
__global__ __launch_bounds__(256)
void contour_raster(const float* __restrict__ loc,    // (4,2,16,16)
                    const float* __restrict__ par,    // (4,12,16,16)
                    unsigned* __restrict__ gmask)     // (4,128,4) u32
{
    __shared__ unsigned bmp[512];   // contour bitmap: row r -> words r*4..r*4+3
    __shared__ unsigned res[512];   // m2 then |= m1
    __shared__ unsigned tmp[512];   // down-scan mask
    const int cell = blockIdx.x;    // 0..1023
    const int b    = cell >> 8;
    const int ij   = cell & 255;
    const int t    = threadIdx.x;

    bmp[t] = 0; bmp[t + 256] = 0;
    __syncthreads();

    const float a0 = loc[((b * 2 + 0) << 8) + ij];
    const float c0 = loc[((b * 2 + 1) << 8) + ij];
    float As[3], Bs[3], Cs[3], Ds[3];
    #pragma unroll
    for (int o = 0; o < 3; ++o) {
        As[o] = par[((b * 12 + o)     << 8) + ij];
        Bs[o] = par[((b * 12 + 3 + o) << 8) + ij];
        Cs[o] = par[((b * 12 + 6 + o) << 8) + ij];
        Ds[o] = par[((b * 12 + 9 + o) << 8) + ij];
    }

    const float TWO_PI = 6.28318530717958647692f;
    for (int idx = t; idx < 10000; idx += 256) {
        float tt = (float)idx * 1e-4f;
        float X = a0, Y = c0;
        #pragma unroll
        for (int o = 0; o < 3; ++o) {
            float s, c;
            sincosf(TWO_PI * tt * (float)(o + 1), &s, &c);
            X = fmaf(As[o], s, fmaf(Bs[o], c, X));
            Y = fmaf(Cs[o], s, fmaf(Ds[o], c, Y));
        }
        int Xi = min(max((int)X, 0), 127);   // trunc-toward-zero then clip
        int Yi = min(max((int)Y, 0), 127);
        atomicOr(&bmp[(Yi << 2) + (Xi >> 5)], 1u << (Xi & 31));
    }
    __syncthreads();

    if (t < 128) {
        // ---- horizontal in/out fill (axis W), 128-bit prefix-XOR scans ----
        unsigned w0 = bmp[(t << 2) + 0], w1 = bmp[(t << 2) + 1];
        unsigned w2 = bmp[(t << 2) + 2], w3 = bmp[(t << 2) + 3];
        unsigned long long L = (unsigned long long)w0 | ((unsigned long long)w1 << 32);
        unsigned long long H = (unsigned long long)w2 | ((unsigned long long)w3 << 32);

        // left scan: G[x] = row[x] & ~row[x-1] (bit0 cleared), inclusive prefix xor
        unsigned long long GlL = L & ~(L << 1);
        unsigned long long GlH = H & ~((H << 1) | (L >> 63));
        GlL &= ~1ull;
        #pragma unroll
        for (int s = 1; s <= 32; s <<= 1) {
            unsigned long long tl = GlL << s;
            unsigned long long th = (GlH << s) | (GlL >> (64 - s));
            GlL ^= tl; GlH ^= th;
        }
        GlH ^= GlL;   // shift-by-64 step

        // right scan: G[x] = row[x] & ~row[x+1] (bit127 cleared), suffix xor
        unsigned long long GrH = H & ~(H >> 1);
        unsigned long long GrL = L & ~((L >> 1) | (H << 63));
        GrH &= ~(1ull << 63);
        #pragma unroll
        for (int s = 1; s <= 32; s <<= 1) {
            unsigned long long tl = (GrL >> s) | (GrH << (64 - s));
            unsigned long long th = GrH >> s;
            GrL ^= tl; GrH ^= th;
        }
        GrL ^= GrH;   // shift-by-64 step

        const unsigned long long epLo = 1ull, epHi = 1ull << 63;
        unsigned long long M2L = ((GlL & GrL) & ~epLo) | (L & epLo);
        unsigned long long M2H = ((GlH & GrH) & ~epHi) | (H & epHi);
        res[(t << 2) + 0] = (unsigned)M2L;
        res[(t << 2) + 1] = (unsigned)(M2L >> 32);
        res[(t << 2) + 2] = (unsigned)M2H;
        res[(t << 2) + 3] = (unsigned)(M2H >> 32);
    } else if (t < 132) {
        // ---- vertical down scan (axis H), 32 columns per word ----
        int w = t - 128;
        unsigned prev = bmp[w];
        tmp[w] = prev;                      // m_down[0] = im[0]
        unsigned pr = 0;
        for (int r = 1; r < 128; ++r) {
            unsigned cur = bmp[(r << 2) + w];
            pr ^= (cur & ~prev);
            tmp[(r << 2) + w] = (r <= 126) ? pr : cur;   // m_down[127] = im[127]
            prev = cur;
        }
    }
    __syncthreads();

    if (t >= 128 && t < 132) {
        // ---- vertical up scan, combine m1 = down & up into res ----
        int w = t - 128;
        unsigned nxt = bmp[(127 << 2) + w];
        res[(127 << 2) + w] |= (tmp[(127 << 2) + w] & nxt);
        unsigned pu = 0;
        for (int r = 126; r >= 0; --r) {
            unsigned cur = bmp[(r << 2) + w];
            pu ^= (cur & ~nxt);
            unsigned mup = (r >= 1) ? pu : cur;          // m_up[0] = im[0]
            res[(r << 2) + w] |= (tmp[(r << 2) + w] & mup);
            nxt = cur;
        }
    }
    __syncthreads();

    atomicOr(&gmask[(b << 9) + t],       res[t]);
    atomicOr(&gmask[(b << 9) + 256 + t], res[t + 256]);
}

// ---------------------------------------------------------------------------
// Kernel 6: bitmask -> float output.
// ---------------------------------------------------------------------------
__global__ void mask_to_float(const unsigned* __restrict__ gmask,
                              float* __restrict__ out)
{
    int p = blockIdx.x * 256 + threadIdx.x;   // 0..65535
    int b = p >> 14;
    int r = (p >> 7) & 127;
    int x = p & 127;
    unsigned wv = gmask[(b << 9) + (r << 2) + (x >> 5)];
    out[p] = ((wv >> (x & 31)) & 1u) ? 1.0f : 0.0f;
}

extern "C" void kernel_launch(void* const* d_in, const int* in_sizes, int n_in,
                              void* d_out, int out_size, void* d_ws, size_t ws_size,
                              hipStream_t stream)
{
    (void)in_sizes; (void)n_in; (void)out_size; (void)ws_size;
    const float* x      = (const float*)d_in[0];
    const float* loc_w1 = (const float*)d_in[1];
    const float* loc_b1 = (const float*)d_in[2];
    const float* loc_g  = (const float*)d_in[3];
    const float* loc_be = (const float*)d_in[4];
    const float* loc_w2 = (const float*)d_in[5];
    const float* loc_b2 = (const float*)d_in[6];
    const float* par_w1 = (const float*)d_in[7];
    const float* par_b1 = (const float*)d_in[8];
    const float* par_g  = (const float*)d_in[9];
    const float* par_be = (const float*)d_in[10];
    const float* par_w2 = (const float*)d_in[11];
    const float* par_b2 = (const float*)d_in[12];

    float* ws       = (float*)d_ws;
    float* y_loc    = ws;               // 65536 f32
    float* y_par    = y_loc + 65536;    // 65536
    float* mean_loc = y_par + 65536;    // 64
    float* rstd_loc = mean_loc + 64;    // 64
    float* mean_par = rstd_loc + 64;    // 64
    float* rstd_par = mean_par + 64;    // 64
    float* loc_out  = rstd_par + 64;    // 2048
    float* par_out  = loc_out + 2048;   // 12288
    unsigned* gmask = (unsigned*)(par_out + 12288);   // 2048 u32

    conv7x7s8_wmma<<<16, 256, 0, stream>>>(x, loc_w1, loc_b1, y_loc);
    conv7x7s8_wmma<<<16, 256, 0, stream>>>(x, par_w1, par_b1, y_par);
    bn_stats<<<64, 256, 0, stream>>>(y_loc, mean_loc, rstd_loc);
    bn_stats<<<64, 256, 0, stream>>>(y_par, mean_par, rstd_par);
    bn_relu_conv1x1<2><<<4, 256, 0, stream>>>(y_loc, mean_loc, rstd_loc, loc_g,
                                              loc_be, loc_w2, loc_b2, loc_out);
    bn_relu_conv1x1<12><<<4, 256, 0, stream>>>(y_par, mean_par, rstd_par, par_g,
                                               par_be, par_w2, par_b2, par_out);
    zero_u32<<<8, 256, 0, stream>>>(gmask, 2048);
    contour_raster<<<1024, 256, 0, stream>>>(loc_out, par_out, gmask);
    mask_to_float<<<256, 256, 0, stream>>>(gmask, (float*)d_out);
}
